// UnetGraphSAGE_11278584119664
// MI455X (gfx1250) — matrix operand order repaired and checked by reading, not compiled, not run on hardware
//
#include <hip/hip_runtime.h>
#include <hip/hip_bf16.h>

// ---------------------------------------------------------------------------
// R-GCN U-Net on cube-sphere graphs, CDNA5 (gfx1250).
//  - Per layer: activations converted once to fp16 [M, Kp] (Kp = K rounded to
//    32, zero-padded); relation weights built fp16 *transposed* [Ncp, Kp]
//    (Ncp = dout rounded to 64, zero-padded).
//  - GEMM loads WMMA fragments DIRECTLY from global memory with per-lane
//    16B-aligned b128 loads (fragment layouts are contiguous runs in the
//    padded buffers) -> no LDS, no barriers, no exec-masked staging.
//    Inner loop = 10 global_load_b128 + 4 v_wmma_f32_16x16x32_f16 per K-step.
//  - Edge aggregation chunked per relation; hx chunk + agg (<=25MB) stay
//    L2-resident (192MB L2); scatters are global_atomic_add_f32.
// ---------------------------------------------------------------------------

typedef _Float16 h8  __attribute__((ext_vector_type(8)));
typedef _Float16 h16 __attribute__((ext_vector_type(16)));
typedef float    f8  __attribute__((ext_vector_type(8)));

#define BM 128
#define BN 64

// C[M x Nc](f32) = A[M x Kp](f16) @ BT[Ncp x Kp](f16, transposed) (+ bias)
// grid: (Ncp/64, M/128), 256 threads = 8 wave32s; wave w -> rows [w*16, w*16+16)
__global__ __launch_bounds__(256)
void gemm_wmma(const _Float16* __restrict__ A, const _Float16* __restrict__ BT,
               const float* __restrict__ bias, float* __restrict__ C,
               int M, int Nc, int Kp, int use_bias) {
  const int m0   = blockIdx.y * BM;
  const int n0   = blockIdx.x * BN;
  const int lane = threadIdx.x & 31;
  const int wave = threadIdx.x >> 5;
  const int hl   = lane & 15;    // lane % 16
  const int hi   = lane >> 4;    // half-wave select

  // A fragment for lane: row = m0+wave*16+hl; halfs K{hi*8..hi*8+7, +16..}
  const _Float16* ap  = A  + (long)(m0 + wave * 16 + hl) * Kp + hi * 8;
  // B fragment j: col = n0+j*16+hl; halfs K{hi*8..hi*8+7, +16..}
  const _Float16* bp0 = BT + (long)(n0 + hl) * Kp + hi * 8;

  const f8 zero = {0.f, 0.f, 0.f, 0.f, 0.f, 0.f, 0.f, 0.f};
  f8 acc[4] = {zero, zero, zero, zero};

  for (int k0 = 0; k0 < Kp; k0 += 32) {
    h8 alo = *(const h8*)(ap + k0);
    h8 ahi = *(const h8*)(ap + k0 + 16);
    h16 a = __builtin_shufflevector(alo, ahi,
                                    0,1,2,3,4,5,6,7,8,9,10,11,12,13,14,15);
#pragma unroll
    for (int j = 0; j < 4; ++j) {
      const _Float16* bp = bp0 + (long)j * 16 * Kp + k0;
      h8 blo = *(const h8*)(bp);
      h8 bhi = *(const h8*)(bp + 16);
      h16 b = __builtin_shufflevector(blo, bhi,
                                      0,1,2,3,4,5,6,7,8,9,10,11,12,13,14,15);
      acc[j] = __builtin_amdgcn_wmma_f32_16x16x32_f16(
          false, a, false, b, (short)0, acc[j], false, false);
    }
  }

  // D layout: lanes 0-15 vgpr i -> M=i; lanes 16-31 -> M=8+i; N = hl
#pragma unroll
  for (int j = 0; j < 4; ++j) {
    int col = n0 + j * 16 + hl;
    if (col >= Nc) continue;
    float bv = use_bias ? bias[col] : 0.f;
#pragma unroll
    for (int i = 0; i < 8; ++i) {
      int row = m0 + wave * 16 + hi * 8 + i;
      C[(long)row * Nc + col] = acc[j][i] + bv;
    }
  }
}

// xh[m, k] = k < K ? f16(x[m, k]) : 0   (Kp-padded)
__global__ void cvt_x_k(const float* __restrict__ x, _Float16* __restrict__ xh,
                        long M, int K, int Kp) {
  long t = (long)blockIdx.x * 256 + threadIdx.x;
  if (t >= M * Kp) return;
  long m = t / Kp;
  int k  = (int)(t - m * Kp);
  xh[t] = (k < K) ? (_Float16)x[m * K + k] : (_Float16)0.f;
}

// WT[r, o, k] = (o<dout && k<din) ? f16(comp[r,0]*bases[0,k,o]+comp[r,1]*bases[1,k,o]) : 0
__global__ void make_wT_k(const float* __restrict__ bases, const float* __restrict__ comp,
                          _Float16* __restrict__ WT, int din, int dout, int Kp, int Ncp) {
  long per = (long)Ncp * Kp;
  long t = (long)blockIdx.x * 256 + threadIdx.x;
  if (t >= 8 * per) return;
  int r = (int)(t / per);
  long rem = t - (long)r * per;
  int o = (int)(rem / Kp), k = (int)(rem - (long)o * Kp);
  float v = 0.f;
  if (o < dout && k < din) {
    long n = (long)din * dout;
    long io = (long)k * dout + o;
    v = comp[r * 2] * bases[io] + comp[r * 2 + 1] * bases[n + io];
  }
  WT[t] = (_Float16)v;
}

// LT[o, k] = (o<dout && k<din) ? f16(loop[k, o]) : 0
__global__ void cvt_loopT_k(const float* __restrict__ loop, _Float16* __restrict__ LT,
                            int din, int dout, int Kp, int Ncp) {
  long t = (long)blockIdx.x * 256 + threadIdx.x;
  if (t >= (long)Ncp * Kp) return;
  int o = (int)(t / Kp), k = (int)(t - (long)o * Kp);
  LT[t] = (o < dout && k < din) ? (_Float16)loop[(long)k * dout + o] : (_Float16)0.f;
}

// One pass per relation: agg[dst] += hx[src] where etype==r. float4 + fp atomics.
__global__ void scatter_rel_k(const float* __restrict__ hx, const int* __restrict__ src,
                              const int* __restrict__ dst, const int* __restrict__ et,
                              float* __restrict__ agg, int E, int dout, int r) {
  int chunks = dout >> 2;
  long t = (long)blockIdx.x * 256 + threadIdx.x;
  if (t >= (long)E * chunks) return;
  int e = (int)(t / chunks);
  int c = (int)(t - (long)e * chunks) * 4;
  if (et[e] != r) return;
  float4 v = *(const float4*)(hx + (long)src[e] * dout + c);
  float* p = agg + (long)dst[e] * dout + c;
  unsafeAtomicAdd(p + 0, v.x);
  unsafeAtomicAdd(p + 1, v.y);
  unsafeAtomicAdd(p + 2, v.z);
  unsafeAtomicAdd(p + 3, v.w);
}

__global__ void relu_k(float* __restrict__ x, long n) {
  long t = (long)blockIdx.x * 256 + threadIdx.x;
  if (t < n) x[t] = fmaxf(x[t], 0.f);
}

// 2x2 mean pool per cube face. out node = f*ro*ro + y2*ro + x2, ro = res/2.
__global__ void meanpool_k(const float* __restrict__ in, float* __restrict__ out,
                           int ro, int C, long total) {
  long t = (long)blockIdx.x * 256 + threadIdx.x;
  if (t >= total) return;
  int c  = (int)(t % C);
  long n = t / C;
  int res = ro * 2;
  int f   = (int)(n / ((long)ro * ro));
  int rem = (int)(n - (long)f * ro * ro);
  int y2 = rem / ro, x2 = rem - y2 * ro;
  long base = ((long)f * res + 2 * y2) * res + 2 * x2;
  float s = in[base * C + c] + in[(base + 1) * C + c] +
            in[(base + res) * C + c] + in[(base + res + 1) * C + c];
  out[t] = 0.25f * s;
}

// ConvTranspose2d(k=2, s=2), PyTorch convention (transpose_kernel=True):
// out[f, co, 2y+dy, 2x+dx] = b[co] + sum_ci in[f, ci, y, x] * w[ci, co, dy, dx]
__global__ void upsample_k(const float* __restrict__ in, const float* __restrict__ w,
                           const float* __restrict__ b, float* __restrict__ out,
                           int res, int Cin, int Cout, long total) {
  long t = (long)blockIdx.x * 256 + threadIdx.x;
  if (t >= total) return;
  int co  = (int)(t % Cout);
  long on = t / Cout;
  int ro  = res * 2;
  int f   = (int)(on / ((long)ro * ro));
  int rem = (int)(on - (long)f * ro * ro);
  int yo = rem / ro, xo = rem - yo * ro;
  int y = yo >> 1, dy = yo & 1, x = xo >> 1, dx = xo & 1;
  long ni = ((long)f * res + y) * res + x;
  const float* ip = in + ni * Cin;
  float acc = b[co];
  for (int ci = 0; ci < Cin; ++ci)
    acc += ip[ci] * w[((ci * Cout + co) * 2 + dy) * 2 + dx];
  out[t] = acc;
}

__global__ void concat_k(const float* __restrict__ a, const float* __restrict__ bb,
                         float* __restrict__ o, long N, int Ca, int Cb) {
  int C = Ca + Cb;
  long t = (long)blockIdx.x * 256 + threadIdx.x;
  if (t >= N * C) return;
  long n = t / C;
  int c  = (int)(t - n * C);
  o[t] = (c < Ca) ? a[n * Ca + c] : bb[n * Cb + (c - Ca)];
}

// ---------------------------------------------------------------------------
// Host side
// ---------------------------------------------------------------------------
static constexpr long N1 = 98304, N2 = 24576, N3 = 6144;

// workspace layout (float units); ~262 MB, lifetime-based reuse
static constexpr long OFF_WT = 0;                   // 8*256*256 halfs
static constexpr long OFF_LT = OFF_WT + 262144;     // 256*256 halfs
static constexpr long OFF_XH = OFF_LT + 32768;      // N1*128 halfs (max Kp input)
static constexpr long OFF_HX = OFF_XH + N1 * 64;    // N1*64 f32 max hx chunk
static constexpr long OFF_A1 = OFF_HX + N1 * 64;    // N1*64
static constexpr long OFF_A2 = OFF_A1 + N1 * 64;    // N1*64  (h22, persists)
static constexpr long OFF_B1 = OFF_A2 + N1 * 64;    // N2*64
static constexpr long OFF_B2 = OFF_B1 + N2 * 64;    // N2*128
static constexpr long OFF_B3 = OFF_B2 + N2 * 128;   // N2*128 (h33, persists)
static constexpr long OFF_C1 = OFF_B3 + N2 * 128;   // N3*128
static constexpr long OFF_C2 = OFF_C1 + N3 * 128;   // N3*256
static constexpr long OFF_C3 = OFF_C2 + N3 * 256;   // N3*256
static constexpr long OFF_D1 = OFF_C3 + N3 * 256;   // N2*128
static constexpr long OFF_D2 = OFF_D1 + N2 * 128;   // N2*256
static constexpr long OFF_E  = OFF_D2 + N2 * 256;   // N1*128
static constexpr long OFF_F1 = OFF_E  + N1 * 128;   // N2*128
static constexpr long OFF_F2 = OFF_F1 + N2 * 128;   // N2*128

struct Ws {
  _Float16 *WT, *LT, *XH;
  float* HX;
};

static void rgc_layer(const float* x, float* out, long N, int din, int dout,
                      const int* src, const int* dst, const int* et, int E,
                      const float* bases, const float* bias, const float* comp,
                      const float* loop, const Ws& w, bool relu, hipStream_t s) {
  const int Kp  = (din + 31) & ~31;
  const int Ncp = ((dout + 63) / 64) * 64;

  long xt = N * Kp;
  cvt_x_k<<<(unsigned)((xt + 255) / 256), 256, 0, s>>>(x, w.XH, N, din, Kp);
  long wt = 8L * Ncp * Kp;
  make_wT_k<<<(unsigned)((wt + 255) / 256), 256, 0, s>>>(bases, comp, w.WT, din, dout, Kp, Ncp);
  long lt = (long)Ncp * Kp;
  cvt_loopT_k<<<(unsigned)((lt + 255) / 256), 256, 0, s>>>(loop, w.LT, din, dout, Kp, Ncp);

  dim3 grid(Ncp / BN, (unsigned)(N / BM));
  // self-loop + bias initializes agg
  gemm_wmma<<<grid, 256, 0, s>>>(w.XH, w.LT, bias, out, (int)N, dout, Kp, 1);

  long sthreads = (long)E * (dout >> 2);
  unsigned sblocks = (unsigned)((sthreads + 255) / 256);
  for (int r = 0; r < 8; ++r) {
    gemm_wmma<<<grid, 256, 0, s>>>(w.XH, w.WT + (long)r * Ncp * Kp, nullptr,
                                   w.HX, (int)N, dout, Kp, 0);
    scatter_rel_k<<<sblocks, 256, 0, s>>>(w.HX, src, dst, et, out, E, dout, r);
  }
  if (relu) {
    long tot = N * dout;
    relu_k<<<(unsigned)((tot + 255) / 256), 256, 0, s>>>(out, tot);
  }
}

extern "C" void kernel_launch(void* const* d_in, const int* in_sizes, int n_in,
                              void* d_out, int out_size, void* d_ws, size_t ws_size,
                              hipStream_t stream) {
  const float* x  = (const float*)d_in[0];
  const int* src1 = (const int*)d_in[1];
  const int* dst1 = (const int*)d_in[2];
  const int* et1  = (const int*)d_in[3];
  const int* src2 = (const int*)d_in[4];
  const int* dst2 = (const int*)d_in[5];
  const int* et2  = (const int*)d_in[6];
  const int* src3 = (const int*)d_in[7];
  const int* dst3 = (const int*)d_in[8];
  const int* et3  = (const int*)d_in[9];
  const int E1 = in_sizes[1], E2 = in_sizes[4], E3 = in_sizes[7];

  // params flattened in jax pytree order (sorted dict keys):
  // c1,c2,c3,c33,c4,c44,c444,c5,c55,c555,c6,c66,c7 each {bases,bias,comp,loop},
  // then up1_b, up1_w, up2_b, up2_w
  auto P = [&](int i) { return (const float*)d_in[i]; };
  const int c1 = 10, c2 = 14, c3 = 18, c33 = 22, c4 = 26, c44 = 30, c444 = 34,
            c5 = 38, c55 = 42, c555 = 46, c6 = 50, c66 = 54, c7 = 58;
  const float* up1_b = P(62); const float* up1_w = P(63);
  const float* up2_b = P(64); const float* up2_w = P(65);

  float* ws = (float*)d_ws;
  Ws W;
  W.WT = (_Float16*)(ws + OFF_WT);
  W.LT = (_Float16*)(ws + OFF_LT);
  W.XH = (_Float16*)(ws + OFF_XH);
  W.HX = ws + OFF_HX;
  float* A1 = ws + OFF_A1;  float* A2 = ws + OFF_A2;
  float* B1 = ws + OFF_B1;  float* B2 = ws + OFF_B2;  float* B3 = ws + OFF_B3;
  float* C1 = ws + OFF_C1;  float* C2 = ws + OFF_C2;  float* C3 = ws + OFF_C3;
  float* D1 = ws + OFF_D1;  float* D2 = ws + OFF_D2;
  float* Ebuf = ws + OFF_E; float* F1 = ws + OFF_F1;  float* F2 = ws + OFF_F2;
  hipStream_t s = stream;

  // ---------------- encoder, level 1 (res 128) ----------------
  rgc_layer(x,  A1, N1, 8,  64, src1, dst1, et1, E1, P(c1), P(c1+1), P(c1+2), P(c1+3), W, true, s);
  rgc_layer(A1, A2, N1, 64, 64, src1, dst1, et1, E1, P(c2), P(c2+1), P(c2+2), P(c2+3), W, true, s); // h22
  meanpool_k<<<(unsigned)((N2 * 64 + 255) / 256), 256, 0, s>>>(A2, B1, 64, 64, N2 * 64);

  // ---------------- level 2 (res 64) ----------------
  rgc_layer(B1, B2, N2, 64, 128, src2, dst2, et2, E2, P(c3),  P(c3+1),  P(c3+2),  P(c3+3),  W, true, s);
  rgc_layer(B2, B3, N2, 128,128, src2, dst2, et2, E2, P(c33), P(c33+1), P(c33+2), P(c33+3), W, true, s); // h33
  meanpool_k<<<(unsigned)((N3 * 128 + 255) / 256), 256, 0, s>>>(B3, C1, 32, 128, N3 * 128);

  // ---------------- bottleneck (res 32) ----------------
  rgc_layer(C1, C2, N3, 128, 256, src3, dst3, et3, E3, P(c4),   P(c4+1),   P(c4+2),   P(c4+3),   W, true, s);
  rgc_layer(C2, C3, N3, 256, 256, src3, dst3, et3, E3, P(c44),  P(c44+1),  P(c44+2),  P(c44+3),  W, true, s);
  rgc_layer(C3, C1, N3, 256, 128, src3, dst3, et3, E3, P(c444), P(c444+1), P(c444+2), P(c444+3), W, true, s);

  // upsample 32 -> 64, Cin=Cout=128; concat with h33 -> [N2, 256]
  upsample_k<<<(unsigned)((N2 * 128 + 255) / 256), 256, 0, s>>>(C1, up1_w, up1_b, D1, 32, 128, 128, N2 * 128);
  concat_k<<<(unsigned)((N2 * 256 + 255) / 256), 256, 0, s>>>(D1, B3, D2, N2, 128, 128);

  // ---------------- decoder, level 2 ----------------
  rgc_layer(D2, F1, N2, 256, 128, src2, dst2, et2, E2, P(c5),   P(c5+1),   P(c5+2),   P(c5+3),   W, true, s);
  rgc_layer(F1, F2, N2, 128, 128, src2, dst2, et2, E2, P(c55),  P(c55+1),  P(c55+2),  P(c55+3),  W, true, s);
  rgc_layer(F2, B1, N2, 128, 64,  src2, dst2, et2, E2, P(c555), P(c555+1), P(c555+2), P(c555+3), W, true, s);

  // upsample 64 -> 128, Cin=Cout=64; concat with h22 -> [N1, 128]
  upsample_k<<<(unsigned)((N1 * 64 + 255) / 256), 256, 0, s>>>(B1, up2_w, up2_b, A1, 64, 64, 64, N1 * 64);
  concat_k<<<(unsigned)((N1 * 128 + 255) / 256), 256, 0, s>>>(A1, A2, Ebuf, N1, 64, 64);

  // ---------------- decoder, level 1 ----------------
  rgc_layer(Ebuf, A1, N1, 128, 64, src1, dst1, et1, E1, P(c6),  P(c6+1),  P(c6+2),  P(c6+3),  W, true, s);
  rgc_layer(A1,   A2, N1, 64,  64, src1, dst1, et1, E1, P(c66), P(c66+1), P(c66+2), P(c66+3), W, true, s);
  rgc_layer(A2, (float*)d_out, N1, 64, 8, src1, dst1, et1, E1,
            P(c7), P(c7+1), P(c7+2), P(c7+3), W, false, s);

  (void)n_in; (void)out_size; (void)ws_size;
}
